// Model_21345987461196
// MI455X (gfx1250) — matrix operand order repaired and checked by reading, not compiled, not run on hardware
//
#include <hip/hip_runtime.h>

typedef __attribute__((ext_vector_type(16))) __bf16 v16bf;
typedef __attribute__((ext_vector_type(8)))  float  v8f;

#define Bc    4
#define Tc    512
#define Nc    64
#define PLc   16
#define STRc  16
#define PNc   33
#define DMc   128
#define DSc   16
#define DFFc  256
#define Ec    8
#define PREDc 96
#define NFc   (DMc*PNc)        // 4224
#define D2c   (DMc*(Ec+1))     // 1152
#define Lpc   (Nc*PNc+1)       // 2113

// ---------------------------------------------------------------------------
// activation codes: 0=none 1=silu 2=gelu(exact) 3=softplus
__device__ __forceinline__ float act_apply(float x, int act) {
  switch (act) {
    case 1: return x / (1.0f + __expf(-x));
    case 2: return 0.5f * x * (1.0f + erff(x * 0.70710678118654752f));
    case 3: return (x > 20.0f) ? x : log1pf(__expf(x));
    default: return x;
  }
}

// ---------------------------------------------------------------------------
// Generic strided-batched GEMM: Y = act(A @ W^T + bias) [* mul] [+ Y_old]
//   A element (m,k)  at A[b*strideA + m*lda_m + k*lda_k]   (fp32)
//   W (Nn,K) row-major (fp32), shared across batch
//   Y (M,Nn) row-major, batch stride strideY
// One wave computes one 16x16 tile via v_wmma_f32_16x16x32_bf16.
// Fast path (interior tile, unit k-stride, K%32==0): unguarded float2 loads,
//   K unrolled by 64 (two WMMAs per iteration) + weight prefetch.
// Edge path: address-clamped unconditional loads, OOB zeroed by 0/1-mask
//   multiplies (not foldable -> loads stay branch-free).
__global__ void gemm_wmma_kernel(
    const float* __restrict__ A, long lda_m, long lda_k, long strideA,
    const float* __restrict__ W,
    const float* __restrict__ bias,
    const float* __restrict__ mul, long strideMul,
    float* __restrict__ Y, long strideY,
    int M, int Nn, int K, int act, int accum)
{
  const int lane = threadIdx.x & 31;
  const int wave = threadIdx.x >> 5;
  const int tilesN = (Nn + 15) >> 4;
  const int tilesM = (M + 15) >> 4;
  const int tile = blockIdx.x * (blockDim.x >> 5) + wave;
  if (tile >= tilesM * tilesN) return;
  const int tm = tile / tilesN;
  const int tn = tile % tilesN;
  const int b  = blockIdx.z;

  const float* Ab = A + (long)b * strideA;
  const float* Mb = mul ? (mul + (long)b * strideMul) : (const float*)0;
  float*       Yb = Y + (long)b * strideY;

  const int lhalf = lane >> 4;
  const int lm    = lane & 15;
  const int mrow  = tm * 16 + lm;   // A row this lane feeds
  const int nrow  = tn * 16 + lm;   // W row this lane feeds
  const bool mok  = (mrow < M);
  const bool nok  = (nrow < Nn);

  v8f acc = {0.f, 0.f, 0.f, 0.f, 0.f, 0.f, 0.f, 0.f};

  // wave-uniform fast-path predicate
  const bool full = (tm * 16 + 16 <= M) && (tn * 16 + 16 <= Nn) &&
                    ((K & 31) == 0) && (lda_k == 1);

  if (full) {
    const float* Ar = Ab + (long)mrow * lda_m;
    const float* Wr = W  + (long)nrow * K;
    const int kaOff = lhalf * 16;   // A lane-half K offset
    const int kbOff = lhalf * 16;   // B lane-half K offset

    // 32-wide slab loader (pairs are K-contiguous in both register layouts)
    auto slab = [&](int k0, v16bf& av, v16bf& bv) {
      float2 a[8], w[8];
#pragma unroll
      for (int v = 0; v < 8; ++v) {
        int kA = k0 + kaOff + 2 * v;
        int kB = k0 + kbOff + 2 * v;
        a[v] = *(const float2*)(Ar + kA);
        w[v] = *(const float2*)(Wr + kB);
      }
#pragma unroll
      for (int v = 0; v < 8; ++v) {
        av[2 * v]     = (__bf16)a[v].x;
        av[2 * v + 1] = (__bf16)a[v].y;
        bv[2 * v]     = (__bf16)w[v].x;
        bv[2 * v + 1] = (__bf16)w[v].y;
      }
    };

    int k0 = 0;
    for (; k0 + 64 <= K; k0 += 64) {
      if (k0 + 128 <= K) {                      // pull next weight slab toward L2/L0
        __builtin_prefetch(Wr + k0 + 64, 0, 3);
        __builtin_prefetch(Ar + k0 + 64, 0, 3);
      }
      v16bf a0, b0, a1, b1;
      slab(k0,      a0, b0);
      slab(k0 + 32, a1, b1);
      acc = __builtin_amdgcn_wmma_f32_16x16x32_bf16(false, a0, false, b0,
                                                    (short)0, acc, false, false);
      acc = __builtin_amdgcn_wmma_f32_16x16x32_bf16(false, a1, false, b1,
                                                    (short)0, acc, false, false);
    }
    for (; k0 < K; k0 += 32) {
      v16bf av, bv;
      slab(k0, av, bv);
      acc = __builtin_amdgcn_wmma_f32_16x16x32_bf16(false, av, false, bv,
                                                    (short)0, acc, false, false);
    }
  } else {
    // clamp rows/cols so every load address is valid; zero OOB by mask-multiply
    const long mA = (long)(mok ? mrow : (M - 1)) * lda_m;
    const long nW = (long)(nok ? nrow : (Nn - 1)) * K;
    const int kmax = K - 1;
    const float mMsk = mok ? 1.f : 0.f;
    const float nMsk = nok ? 1.f : 0.f;
    for (int k0 = 0; k0 < K; k0 += 32) {
      v16bf av, bv;
#pragma unroll
      for (int v = 0; v < 8; ++v) {
        int kA = k0 + lhalf * 16 + 2 * v;
        int c0 = (kA     < K) ? kA     : kmax;
        int c1 = (kA + 1 < K) ? kA + 1 : kmax;
        float ka0 = (kA     < K) ? mMsk : 0.f;
        float ka1 = (kA + 1 < K) ? mMsk : 0.f;
        float a0 = Ab[mA + (long)c0 * lda_k] * ka0;   // 0*x not foldable: load stays
        float a1 = Ab[mA + (long)c1 * lda_k] * ka1;
        int kB = k0 + lhalf * 16 + 2 * v;
        int d0 = (kB     < K) ? kB     : kmax;
        int d1 = (kB + 1 < K) ? kB + 1 : kmax;
        float kb0 = (kB     < K) ? nMsk : 0.f;
        float kb1 = (kB + 1 < K) ? nMsk : 0.f;
        float w0 = W[nW + d0] * kb0;
        float w1 = W[nW + d1] * kb1;
        av[2 * v]     = (__bf16)a0;
        av[2 * v + 1] = (__bf16)a1;
        bv[2 * v]     = (__bf16)w0;
        bv[2 * v + 1] = (__bf16)w1;
      }
      acc = __builtin_amdgcn_wmma_f32_16x16x32_bf16(false, av, false, bv,
                                                    (short)0, acc, false, false);
    }
  }

#pragma unroll
  for (int r = 0; r < 8; ++r) {
    int mo = tm * 16 + r + lhalf * 8;
    int no = tn * 16 + lm;
    if (mo < M && no < Nn) {
      float v = acc[r];
      if (bias) v += bias[no];
      v = act_apply(v, act);
      long idx = (long)mo * Nn + no;
      if (Mb) v *= Mb[idx];
      if (accum) v += Yb[idx];
      Yb[idx] = v;
    }
  }
}

// ---------------------------------------------------------------------------
// per-(b,n) mean / std over T
__global__ void stats_kernel(const float* __restrict__ z, float* meanb, float* stdb)
{
  int bn = blockIdx.x; int b = bn / Nc; int n = bn % Nc;
  __shared__ float r1[256], r2[256];
  int tid = threadIdx.x;
  float s = 0.f, sq = 0.f;
  for (int t = tid; t < Tc; t += blockDim.x) {
    float v = z[((long)b * Tc + t) * Nc + n];
    s += v; sq += v * v;
  }
  r1[tid] = s; r2[tid] = sq; __syncthreads();
  for (int st = blockDim.x >> 1; st > 0; st >>= 1) {
    if (tid < st) { r1[tid] += r1[tid + st]; r2[tid] += r2[tid + st]; }
    __syncthreads();
  }
  if (tid == 0) {
    float m = r1[0] / (float)Tc;
    float var = r2[0] / (float)Tc - m * m;
    meanb[bn] = m;
    stdb[bn]  = sqrtf(var + 1e-5f);
  }
}

__global__ void normalize_kernel(const float* __restrict__ z, const float* meanb,
                                 const float* stdb, float* zn, long n)
{
  long i = (long)blockIdx.x * blockDim.x + threadIdx.x;
  if (i >= n) return;
  int nn = (int)(i % Nc);
  int b  = (int)(i / ((long)Tc * Nc));
  zn[i] = (z[i] - meanb[b * Nc + nn]) / stdb[b * Nc + nn];
}

__global__ void patch_kernel(const float* __restrict__ zn, float* P, long n)
{
  long i = (long)blockIdx.x * blockDim.x + threadIdx.x;
  if (i >= n) return;
  int pl = (int)(i % PLc); long r = i / PLc;
  int pn = (int)(r % PNc); r /= PNc;
  int nn = (int)(r % Nc);  int b = (int)(r / Nc);
  int t = pn * STRc + pl; if (t >= Tc) t = Tc - 1;  // replicated tail column
  P[i] = zn[((long)b * Tc + t) * Nc + nn];
}

// residual + layernorm: Y = LN(X [+ R]) * g + b  (rows of width D, in-place safe)
__global__ void resln_kernel(const float* __restrict__ X, const float* __restrict__ R,
                             const float* __restrict__ g, const float* __restrict__ bt,
                             float* __restrict__ Y, int D)
{
  __shared__ float srow[1280];
  __shared__ float red[256];
  long row = blockIdx.x;
  const float* x = X + row * D;
  const float* r = R ? (R + row * D) : (const float*)0;
  int tid = threadIdx.x;
  float s = 0.f;
  for (int i = tid; i < D; i += blockDim.x) {
    float v = x[i] + (r ? r[i] : 0.f);
    srow[i] = v; s += v;
  }
  red[tid] = s; __syncthreads();
  for (int st = blockDim.x >> 1; st > 0; st >>= 1) {
    if (tid < st) red[tid] += red[tid + st];
    __syncthreads();
  }
  float mean = red[0] / (float)D; __syncthreads();
  float vs = 0.f;
  for (int i = tid; i < D; i += blockDim.x) { float d = srow[i] - mean; vs += d * d; }
  red[tid] = vs; __syncthreads();
  for (int st = blockDim.x >> 1; st > 0; st >>= 1) {
    if (tid < st) red[tid] += red[tid + st];
    __syncthreads();
  }
  float inv = rsqrtf(red[0] / (float)D + 1e-5f);
  for (int i = tid; i < D; i += blockDim.x)
    Y[row * D + i] = (srow[i] - mean) * inv * g[i] + bt[i];
}

// depthwise causal conv(k=2) + SiLU; rev flips the time direction
__global__ void conv_silu_kernel(const float* __restrict__ xz, float* __restrict__ xc,
                                 const float* __restrict__ cw, const float* __restrict__ cb,
                                 int L, int D, int rev, long n)
{
  long i = (long)blockIdx.x * blockDim.x + threadIdx.x;
  if (i >= n) return;
  int d = (int)(i % D);
  int t = (int)((i / D) % L);
  int b = (int)(i / ((long)D * L));
  float xi = xz[((long)b * L + t) * (2 * D) + d];
  int tp = rev ? (t + 1) : (t - 1);
  float xp = (tp >= 0 && tp < L) ? xz[((long)b * L + tp) * (2 * D) + d] : 0.f;
  float v = xi * cw[2 * d + 1] + xp * cw[2 * d + 0] + cb[d];
  xc[((long)b * L + t) * D + d] = v / (1.f + __expf(-v));
}

// selective-scan: h[n] = exp(dt*A[d][n])*h[n] + dt*B_t[n]*xc ; y = <h,C_t>
// ym = (y + xc*D) * silu(zg)
__global__ void ssm_scan_kernel(const float* __restrict__ proj, int projW, int dtr,
                                const float* __restrict__ dt, const float* __restrict__ xc,
                                const float* __restrict__ xz,
                                const float* __restrict__ Al, const float* __restrict__ Dp,
                                float* __restrict__ ym, int L, int D, int rev)
{
  int b = blockIdx.x;
  int d = blockIdx.y * blockDim.x + threadIdx.x;
  __shared__ float sB[DSc], sC[DSc];
  float h[DSc], a[DSc];
  if (d < D) {
#pragma unroll
    for (int n = 0; n < DSc; ++n) { h[n] = 0.f; a[n] = -__expf(Al[(long)d * DSc + n]); }
  }
  for (int s = 0; s < L; ++s) {
    int t = rev ? (L - 1 - s) : s;
    const float* pr = proj + ((long)b * L + t) * projW;
    if (threadIdx.x < DSc)            sB[threadIdx.x]        = pr[dtr + threadIdx.x];
    else if (threadIdx.x < 2 * DSc)   sC[threadIdx.x - DSc]  = pr[dtr + threadIdx.x];
    __syncthreads();
    if (d < D) {
      long o = ((long)b * L + t) * D + d;
      float dtv = dt[o];
      float xcv = xc[o];
      float y = 0.f;
#pragma unroll
      for (int n = 0; n < DSc; ++n) {
        h[n] = __expf(dtv * a[n]) * h[n] + dtv * sB[n] * xcv;
        y += h[n] * sC[n];
      }
      float zg = xz[((long)b * L + t) * (2 * D) + D + d];
      ym[o] = (y + xcv * Dp[d]) * (zg / (1.f + __expf(-zg)));
    }
    __syncthreads();
  }
}

// f[b,n, d*PN+pn] = u[b, n*PN+pn, d]
__global__ void fextract_kernel(const float* __restrict__ u, float* __restrict__ f, long n)
{
  long i = (long)blockIdx.x * blockDim.x + threadIdx.x;
  if (i >= n) return;
  int pn = (int)(i % PNc); long r = i / PNc;
  int d  = (int)(r % DMc); r /= DMc;
  int nn = (int)(r % Nc);  int b = (int)(r / Nc);
  f[i] = u[((long)b * Lpc + nn * PNc + pn) * DMc + d];
}

// logits = fh @ emb_w^T + emb_b ; bmax = argmax+1 ;
// V = [nview | (e<bmax ? fh*tok[e] : 0), e=0..E-1]
__global__ void assemble_v_kernel(const float* __restrict__ fh, const float* __restrict__ nview,
                                  const float* __restrict__ emb_w, const float* __restrict__ emb_b,
                                  const float* __restrict__ tok, float* __restrict__ V)
{
  int bn = blockIdx.x;
  __shared__ float row[DMc];
  __shared__ float logits[Ec];
  __shared__ int bmax;
  int tid = threadIdx.x;
  if (tid < DMc) row[tid] = fh[(long)bn * DMc + tid];
  __syncthreads();
  if (tid < Ec) {
    float s = emb_b[tid];
    for (int d = 0; d < DMc; ++d) s += row[d] * emb_w[tid * DMc + d];
    logits[tid] = s;
  }
  __syncthreads();
  if (tid == 0) {
    int arg = 0; float best = logits[0];
    for (int e = 1; e < Ec; ++e) if (logits[e] > best) { best = logits[e]; arg = e; }
    bmax = arg + 1;
  }
  __syncthreads();
  if (tid < DMc) {
    long base = (long)bn * D2c;
    V[base + tid] = nview[(long)bn * DMc + tid];
    for (int e = 0; e < Ec; ++e)
      V[base + (e + 1) * DMc + tid] = (e < bmax) ? row[tid] * tok[e] : 0.f;
  }
}

// out[b,p,n] = zf[b,n,p] * std[b,n] + mean[b,n]
__global__ void final_kernel(const float* __restrict__ zf, const float* meanb,
                             const float* stdb, float* __restrict__ out, long n)
{
  long i = (long)blockIdx.x * blockDim.x + threadIdx.x;
  if (i >= n) return;
  int nn = (int)(i % Nc);
  int p  = (int)((i / Nc) % PREDc);
  int b  = (int)(i / ((long)Nc * PREDc));
  out[i] = zf[((long)b * Nc + nn) * PREDc + p] * stdb[b * Nc + nn] + meanb[b * Nc + nn];
}

// ---------------------------------------------------------------------------
// host-side orchestration
static inline void gemm(hipStream_t st,
    const float* A, long lda_m, long lda_k, long sA,
    const float* W, const float* bias,
    const float* mul, long sMul,
    float* Y, long sY,
    int batch, int M, int Nn, int K, int act, int accum)
{
  int tiles = ((M + 15) >> 4) * ((Nn + 15) >> 4);
  dim3 grid((unsigned)((tiles + 3) / 4), 1, (unsigned)batch);
  gemm_wmma_kernel<<<grid, dim3(128, 1, 1), 0, st>>>(A, lda_m, lda_k, sA, W, bias,
                                                     mul, sMul, Y, sY, M, Nn, K, act, accum);
}

static inline dim3 ewgrid(long n) { return dim3((unsigned)((n + 255) / 256)); }

struct MP { const float *inp,*cw,*cb,*xp,*dtw,*dtb,*Al,*Dp,*op; };
struct Scr { float *xz,*xc,*proj,*dt,*ym,*nw,*ffn1; };

static MP getMP(void* const* d_in, int base) {
  MP m;
  m.inp = (const float*)d_in[base + 0];
  m.cw  = (const float*)d_in[base + 1];
  m.cb  = (const float*)d_in[base + 2];
  m.xp  = (const float*)d_in[base + 3];
  m.dtw = (const float*)d_in[base + 4];
  m.dtb = (const float*)d_in[base + 5];
  m.Al  = (const float*)d_in[base + 6];
  m.Dp  = (const float*)d_in[base + 7];
  m.op  = (const float*)d_in[base + 8];
  return m;
}

static void run_mamba(hipStream_t st, const Scr& s, const MP& m,
                      const float* x, int L, int D, int dtr, int rev, int acc)
{
  int projW = dtr + 2 * DSc;
  long LD = (long)L * D;
  // xz = x @ in_proj^T                        (B,L,2D)
  gemm(st, x, D, 1, LD, m.inp, 0, 0, 0, s.xz, LD * 2, Bc, L, 2 * D, D, 0, 0);
  // xc = silu(depthwise-conv(xi))             (B,L,D)
  long n = (long)Bc * L * D;
  conv_silu_kernel<<<ewgrid(n), 256, 0, st>>>(s.xz, s.xc, m.cw, m.cb, L, D, rev, n);
  // proj = xc @ x_proj^T                      (B,L,dtr+2*DS)
  gemm(st, s.xc, D, 1, LD, m.xp, 0, 0, 0, s.proj, (long)L * projW, Bc, L, projW, D, 0, 0);
  // dt = softplus(dtl @ dt_w^T + dt_b)        (B,L,D)
  gemm(st, s.proj, projW, 1, (long)L * projW, m.dtw, m.dtb, 0, 0, s.dt, LD, Bc, L, D, dtr, 3, 0);
  // selective scan -> ym                      (B,L,D)
  ssm_scan_kernel<<<dim3(Bc, (unsigned)((D + 127) / 128)), 128, 0, st>>>(
      s.proj, projW, dtr, s.dt, s.xc, s.xz, m.Al, m.Dp, s.ym, L, D, rev);
  // new (+)= ym @ out_proj^T
  gemm(st, s.ym, D, 1, LD, m.op, 0, 0, 0, s.nw, LD, Bc, L, D, D, 0, acc);
}

static void run_layer(hipStream_t st, const Scr& s, void* const* d_in, int base,
                      float* x, int L, int D, int dtr, int dff)
{
  MP mf = getMP(d_in, base + 0);
  MP mb = getMP(d_in, base + 9);
  const float* w1  = (const float*)d_in[base + 18];
  const float* b1  = (const float*)d_in[base + 19];
  const float* w2  = (const float*)d_in[base + 20];
  const float* b2  = (const float*)d_in[base + 21];
  const float* l1g = (const float*)d_in[base + 22];
  const float* l1b = (const float*)d_in[base + 23];
  const float* l2g = (const float*)d_in[base + 24];
  const float* l2b = (const float*)d_in[base + 25];
  long LD = (long)L * D;

  run_mamba(st, s, mf, x, L, D, dtr, 0, 0);
  run_mamba(st, s, mb, x, L, D, dtr, 1, 1);
  resln_kernel<<<Bc * L, 256, 0, st>>>(x, s.nw, l1g, l1b, x, D);
  gemm(st, x, D, 1, LD, w1, b1, 0, 0, s.ffn1, (long)L * dff, Bc, L, dff, D, 2, 0);
  gemm(st, s.ffn1, dff, 1, (long)L * dff, w2, b2, 0, 0, s.nw, LD, Bc, L, D, dff, 0, 0);
  resln_kernel<<<Bc * L, 256, 0, st>>>(x, s.nw, l2g, l2b, x, D);
}

// param index map (setup_inputs insertion order, recursively flattened)
enum {
  IN_Z = 0,
  P_WP_W = 4, P_WP_B = 5,
  P_PENC0 = 6,                       // 26 leaves per layer
  P_PENC_NG = 58, P_PENC_NB = 59,
  P_HGATE_W = 60, P_HGATE_B = 61,
  P_HLIN_W = 62, P_HLIN_B = 63,
  P_PFFN_W = 64, P_PFFN_B = 65,
  P_TOK = 66,
  P_EMB_W = 67, P_EMB_B = 68,
  P_LIN2_W = 69, P_LIN2_B = 70,
  P_GATE2_W = 71, P_GATE2_B = 72,
  P_NVH_W = 73, P_NVH_B = 74,
  P_VIEW1_W = 75, P_VIEW1_B = 76,
  P_VIEW2_W = 77, P_VIEW2_B = 78,
  P_VIEW3_W = 79, P_VIEW3_B = 80,
  P_ENC0 = 81,
  P_ENC_NG = 133, P_ENC_NB = 134,
};

extern "C" void kernel_launch(void* const* d_in, const int* in_sizes, int n_in,
                              void* d_out, int out_size, void* d_ws, size_t ws_size,
                              hipStream_t stream)
{
  (void)in_sizes; (void)n_in; (void)out_size; (void)ws_size;
  auto F = [&](int i) -> const float* { return (const float*)d_in[i]; };
  const float* z = F(IN_Z);

  float* ws = (float*)d_ws;
  size_t off = 0;
  auto alloc = [&](long nelem) -> float* { float* p = ws + off; off += (size_t)nelem; return p; };

  float* zn    = alloc((long)Bc * Tc * Nc);
  float* meanb = alloc(Bc * Nc);
  float* stdb  = alloc(Bc * Nc);
  float* pv    = alloc((long)Bc * Tc);
  float* patch = alloc((long)Bc * Nc * PNc * PLc);
  float* u     = alloc((long)Bc * Lpc * DMc);
  float* nview = alloc((long)Bc * Nc * DMc);

  Scr s;
  long mxLD  = (long)Lpc * DMc;  if ((long)Nc * D2c  > mxLD)  mxLD  = (long)Nc * D2c;
  long mxL2D = 2 * (long)Lpc * DMc; if (2L * Nc * D2c > mxL2D) mxL2D = 2L * Nc * D2c;
  long mxPRJ = (long)Lpc * (8 + 2 * DSc); if ((long)Nc * (72 + 2 * DSc) > mxPRJ) mxPRJ = (long)Nc * (72 + 2 * DSc);
  long mxFF  = (long)Lpc * DFFc; if ((long)Nc * DFFc > mxFF) mxFF = (long)Nc * DFFc;
  s.xz   = alloc(Bc * mxL2D);
  s.xc   = alloc(Bc * mxLD);
  s.proj = alloc(Bc * mxPRJ);
  s.dt   = alloc(Bc * mxLD);
  s.ym   = alloc(Bc * mxLD);
  s.nw   = alloc(Bc * mxLD);
  s.ffn1 = alloc(Bc * mxFF);

  float* fb  = alloc((long)Bc * Nc * NFc);
  float* gb  = alloc((long)Bc * Nc * NFc);
  float* fh  = alloc((long)Bc * Nc * DMc);
  float* vA  = alloc((long)Bc * Nc * D2c);
  float* vB  = alloc((long)Bc * Nc * D2c);
  float* g2  = alloc((long)Bc * Nc * D2c);
  float* zf  = alloc((long)Bc * Nc * PREDc);

  // ---- instance-norm over T ----
  stats_kernel<<<Bc * Nc, 256, 0, stream>>>(z, meanb, stdb);
  long nz = (long)Bc * Tc * Nc;
  normalize_kernel<<<ewgrid(nz), 256, 0, stream>>>(z, meanb, stdb, zn, nz);

  // ---- pview: pv = zn @ view2^T ; u[:,2112,:] = pv @ view1^T ----
  gemm(stream, zn, Nc, 1, (long)Tc * Nc, F(P_VIEW2_W), F(P_VIEW2_B), 0, 0,
       pv, Tc, Bc, Tc, 1, Nc, 0, 0);
  gemm(stream, pv, Tc, 1, Tc, F(P_VIEW1_W), F(P_VIEW1_B), 0, 0,
       u + (long)(Lpc - 1) * DMc, (long)Lpc * DMc, Bc, 1, DMc, Tc, 0, 0);

  // ---- nview = zn^T @ view3^T ----
  gemm(stream, zn, 1, Nc, (long)Tc * Nc, F(P_VIEW3_W), F(P_VIEW3_B), 0, 0,
       nview, (long)Nc * DMc, Bc, Nc, DMc, Tc, 0, 0);

  // ---- patchify + W_P: u[:, :2112, :] ----
  long np = (long)Bc * Nc * PNc * PLc;
  patch_kernel<<<ewgrid(np), 256, 0, stream>>>(zn, patch, np);
  gemm(stream, patch, PLc, 1, (long)Nc * PNc * PLc, F(P_WP_W), F(P_WP_B), 0, 0,
       u, (long)Lpc * DMc, Bc, Nc * PNc, DMc, PLc, 0, 0);

  // ---- patch encoder: 2 bi-Mamba layers over L=2113, D=128 ----
  for (int l = 0; l < 2; ++l)
    run_layer(stream, s, d_in, P_PENC0 + 26 * l, u, Lpc, DMc, 8, DFFc);
  resln_kernel<<<Bc * Lpc, 256, 0, stream>>>(u, (const float*)0, F(P_PENC_NG), F(P_PENC_NB), u, DMc);

  // ---- head: f, g = silu(f@hg^T+b)*f, fh = g@hl^T+b ----
  long nf = (long)Bc * Nc * NFc;
  fextract_kernel<<<ewgrid(nf), 256, 0, stream>>>(u, fb, nf);
  gemm(stream, fb, NFc, 1, 0, F(P_HGATE_W), F(P_HGATE_B), fb, 0,
       gb, 0, 1, Bc * Nc, NFc, NFc, 1, 0);
  gemm(stream, gb, NFc, 1, 0, F(P_HLIN_W), F(P_HLIN_B), 0, 0,
       fh, 0, 1, Bc * Nc, DMc, NFc, 0, 0);

  // ---- token expansion + argmax masking + concat nview -> vA (B,N,D2) ----
  assemble_v_kernel<<<Bc * Nc, 128, 0, stream>>>(fh, nview, F(P_EMB_W), F(P_EMB_B), F(P_TOK), vA);

  // ---- vB = silu(vA @ lin2^T + b) ----
  gemm(stream, vA, D2c, 1, 0, F(P_LIN2_W), F(P_LIN2_B), 0, 0,
       vB, 0, 1, Bc * Nc, D2c, D2c, 1, 0);

  // ---- outer encoder: 2 bi-Mamba layers over L=64, D=1152 ----
  for (int l = 0; l < 2; ++l)
    run_layer(stream, s, d_in, P_ENC0 + 26 * l, vB, Nc, D2c, 72, DFFc);
  resln_kernel<<<Bc * Nc, 256, 0, stream>>>(vB, (const float*)0, F(P_ENC_NG), F(P_ENC_NB), vB, D2c);

  // ---- g2 = silu(vB @ gate2^T + b) * vB ; zf = g2 @ nvh^T + b ----
  gemm(stream, vB, D2c, 1, 0, F(P_GATE2_W), F(P_GATE2_B), vB, 0,
       g2, 0, 1, Bc * Nc, D2c, D2c, 1, 0);
  gemm(stream, g2, D2c, 1, 0, F(P_NVH_W), F(P_NVH_B), 0, 0,
       zf, 0, 1, Bc * Nc, PREDc, D2c, 0, 0);
  // zf += fh @ pffn^T + b
  gemm(stream, fh, DMc, 1, 0, F(P_PFFN_W), F(P_PFFN_B), 0, 0,
       zf, 0, 1, Bc * Nc, PREDc, DMc, 0, 1);

  // ---- transpose + de-normalize ----
  long no = (long)Bc * PREDc * Nc;
  final_kernel<<<ewgrid(no), 256, 0, stream>>>(zf, meanb, stdb, (float*)d_out, no);
}
